// TPQwenAttention_39470749450829
// MI455X (gfx1250) — matrix operand order repaired and checked by reading, not compiled, not run on hardware
//
#include <hip/hip_runtime.h>

#define H    32
#define KVH  8
#define Dh   128
#define Bsz  2
#define Ssz  2048
#define Esz  4096
#define Mrows (Bsz*Ssz)              // 4096 tokens
#define QKVN  (H*Dh + 2*KVH*Dh)      // 6144

typedef __attribute__((ext_vector_type(16))) __bf16 v16bf;
typedef __attribute__((ext_vector_type(8)))  float  v8f;

union FragBF {
    v16bf v;
    uint4 q[2];
    unsigned short us[16];
};

__device__ __forceinline__ unsigned short f2bf(float f) {
    unsigned int u = __float_as_uint(f);
    u += 0x7FFFu + ((u >> 16) & 1u);           // round-to-nearest-even
    return (unsigned short)(u >> 16);
}
__device__ __forceinline__ float bf2f(unsigned short h) {
    return __uint_as_float(((unsigned int)h) << 16);
}

// ---------------------------------------------------------------- cvt f32->bf16
__global__ void cvt_f32_bf16(const float* __restrict__ in,
                             unsigned short* __restrict__ out, int n) {
    int i = blockIdx.x * blockDim.x + threadIdx.x;
    if (i < n) out[i] = f2bf(in[i]);
}

// ---------------------------------------------------------------- WMMA GEMM
// C[M,N] = A[M,K](bf16) * B[K,N](bf16) + bias. 128x128 block tile, 8 waves,
// LDS double-buffered (one barrier per k-step, global loads overlap WMMA).
template <bool OUT_BF16>
__global__ __launch_bounds__(256) void gemm_bf16_wmma(
    const unsigned short* __restrict__ A,
    const unsigned short* __restrict__ Bw,
    const float* __restrict__ bias,          // may be nullptr
    void* __restrict__ outp,
    int Mi, int Ni, int Ki, int ldo, int col_off)
{
    __shared__ unsigned short As[2][128 * 32];  // [m][k] row-major
    __shared__ unsigned short Bs[2][128 * 32];  // [n][k] (k-transposed)

    const int tid  = threadIdx.x;
    const int lane = tid & 31;
    const int wave = tid >> 5;
    const int half = lane >> 4;
    const int l16  = lane & 15;
    const int wm   = wave >> 1;              // 0..3 -> 32 rows each
    const int wn   = wave & 1;               // 0..1 -> 64 cols each
    const int m0   = blockIdx.y * 128;
    const int n0   = blockIdx.x * 128;

    v8f acc[2][4];
    #pragma unroll
    for (int mi = 0; mi < 2; ++mi)
        #pragma unroll
        for (int ni = 0; ni < 4; ++ni)
            acc[mi][ni] = (v8f){0.f,0.f,0.f,0.f,0.f,0.f,0.f,0.f};

    const int arow = tid >> 1;               // 0..127
    const int akh  = (tid & 1) * 16;         // 0 / 16
    const int bk   = tid >> 3;               // 0..31
    const int bn   = (tid & 7) * 16;         // 0..112

    uint4 a0, a1;
    union { uint4 q[2]; unsigned short us[16]; } tb;

    // prologue: stage tile 0
    {
        const uint4* ga = (const uint4*)(A + (size_t)(m0 + arow) * Ki + akh);
        a0 = ga[0]; a1 = ga[1];
        const uint4* gb = (const uint4*)(Bw + (size_t)bk * Ni + n0 + bn);
        tb.q[0] = gb[0]; tb.q[1] = gb[1];
        *(uint4*)(&As[0][arow * 32 + akh])     = a0;
        *(uint4*)(&As[0][arow * 32 + akh + 8]) = a1;
        #pragma unroll
        for (int i = 0; i < 16; ++i) Bs[0][(bn + i) * 32 + bk] = tb.us[i];
    }
    __syncthreads();

    const int nk = Ki >> 5;
    int p = 0;
    for (int kt = 0; kt < nk; ++kt) {
        const bool more = (kt + 1) < nk;
        if (more) {
            const int k0 = (kt + 1) * 32;
            const uint4* ga = (const uint4*)(A + (size_t)(m0 + arow) * Ki + k0 + akh);
            a0 = ga[0]; a1 = ga[1];
            const uint4* gb = (const uint4*)(Bw + (size_t)(k0 + bk) * Ni + n0 + bn);
            tb.q[0] = gb[0]; tb.q[1] = gb[1];
        }

        // fragments from buffer p
        FragBF af[2], bf[4];
        #pragma unroll
        for (int mi = 0; mi < 2; ++mi) {
            int m = wm * 32 + mi * 16 + l16;
            af[mi].q[0] = *(const uint4*)(&As[p][m * 32 + half * 8]);       // K 0-7 / 8-15
            af[mi].q[1] = *(const uint4*)(&As[p][m * 32 + 16 + half * 8]);  // K 16-23 / 24-31
        }
        #pragma unroll
        for (int ni = 0; ni < 4; ++ni) {
            int n = wn * 64 + ni * 16 + l16;
            bf[ni].q[0] = *(const uint4*)(&Bs[p][n * 32 + half * 8]);       // K 0-7 / 8-15
            bf[ni].q[1] = *(const uint4*)(&Bs[p][n * 32 + 16 + half * 8]);
        }
        #pragma unroll
        for (int mi = 0; mi < 2; ++mi)
            #pragma unroll
            for (int ni = 0; ni < 4; ++ni)
                acc[mi][ni] = __builtin_amdgcn_wmma_f32_16x16x32_bf16(
                    false, af[mi].v, false, bf[ni].v, (short)0, acc[mi][ni], false, false);

        if (more) {
            // stage next tile into the other buffer (no reader conflict)
            *(uint4*)(&As[p ^ 1][arow * 32 + akh])     = a0;
            *(uint4*)(&As[p ^ 1][arow * 32 + akh + 8]) = a1;
            #pragma unroll
            for (int i = 0; i < 16; ++i) Bs[p ^ 1][(bn + i) * 32 + bk] = tb.us[i];
        }
        __syncthreads();
        p ^= 1;
    }

    #pragma unroll
    for (int mi = 0; mi < 2; ++mi)
        #pragma unroll
        for (int ni = 0; ni < 4; ++ni) {
            int ncol = n0 + wn * 64 + ni * 16 + l16;
            float bv = bias ? bias[ncol] : 0.f;
            #pragma unroll
            for (int r = 0; r < 8; ++r) {
                int mrow = m0 + wm * 32 + mi * 16 + half * 8 + r;
                float vv = acc[mi][ni][r] + bv;
                if (OUT_BF16)
                    ((unsigned short*)outp)[(size_t)mrow * ldo + col_off + ncol] = f2bf(vv);
                else
                    ((float*)outp)[(size_t)mrow * ldo + col_off + ncol] = vv;
            }
        }
}

// ---------------------------------------------------------------- RoPE + split
__global__ void rope_split(const unsigned short* __restrict__ qkv,
                           unsigned short* __restrict__ Q,
                           unsigned short* __restrict__ Kd,
                           unsigned short* __restrict__ Vd)
{
    const int per_tok = (H + 2 * KVH) * 64;  // 3072
    int i = blockIdx.x * blockDim.x + threadIdx.x;
    if (i >= Bsz * Ssz * per_tok) return;
    int tok  = i / per_tok;
    int w    = i - tok * per_tok;
    int head = w >> 6;
    int d    = w & 63;
    int b    = tok / Ssz, s = tok - b * Ssz;

    const unsigned short* src;
    if (head < H)            src = qkv + (size_t)tok * QKVN + head * Dh;
    else if (head < H + KVH) src = qkv + (size_t)tok * QKVN + Esz + (head - H) * Dh;
    else                     src = qkv + (size_t)tok * QKVN + Esz + KVH * Dh + (head - H - KVH) * Dh;

    if (head < H + KVH) {
        float x1 = bf2f(src[d]), x2 = bf2f(src[d + 64]);
        float inv = __expf(-(float)d * 0.14391156831212787f);  // ln(10000)/64
        float ang = (float)s * inv;
        float sn, cs;
        __sincosf(ang, &sn, &cs);
        float o1 = x1 * cs - x2 * sn;
        float o2 = x2 * cs + x1 * sn;
        unsigned short* dst = (head < H)
            ? Q  + ((size_t)(b * H   + head)        * Ssz + s) * Dh
            : Kd + ((size_t)(b * KVH + (head - H))  * Ssz + s) * Dh;
        dst[d]      = f2bf(o1);
        dst[d + 64] = f2bf(o2);
    } else {
        unsigned short* dst = Vd + ((size_t)(b * KVH + (head - H - KVH)) * Ssz + s) * Dh;
        dst[d]      = src[d];
        dst[d + 64] = src[d + 64];
    }
}

// ---------------------------------------------------------------- flash attention
// 4 waves, each owns 16 q-rows; k-tiles of 32; causal; online softmax.
__global__ __launch_bounds__(128) void attn_fa(
    const unsigned short* __restrict__ Q,
    const unsigned short* __restrict__ K,
    const unsigned short* __restrict__ V,
    unsigned short* __restrict__ O)
{
    __shared__ unsigned short Vt[Dh * 32];       // [d][k_local]  8 KB
    __shared__ unsigned short Pl[4 * 16 * 32];   // per-wave P    4 KB

    const int tid  = threadIdx.x;
    const int lane = tid & 31;
    const int wave = tid >> 5;
    const int half = lane >> 4;
    const int l16  = lane & 15;

    const int b   = blockIdx.z;
    const int h   = blockIdx.y;
    const int kvh = h / (H / KVH);
    const int q0  = blockIdx.x * 64;
    const int qw  = q0 + wave * 16;

    const unsigned short* Qb = Q + ((size_t)(b * H   + h)   * Ssz) * Dh;
    const unsigned short* Kb = K + ((size_t)(b * KVH + kvh) * Ssz) * Dh;
    const unsigned short* Vb = V + ((size_t)(b * KVH + kvh) * Ssz) * Dh;

    // resident Q fragments, one per 32-wide K-dim chain step
    FragBF qa[4];
    {
        const unsigned short* qr = Qb + (size_t)(qw + l16) * Dh;
        #pragma unroll
        for (int kk = 0; kk < 4; ++kk) {
            qa[kk].q[0] = *(const uint4*)(qr + kk * 32 + half * 8);
            qa[kk].q[1] = *(const uint4*)(qr + kk * 32 + 16 + half * 8);
        }
    }

    v8f o[8];
    #pragma unroll
    for (int c = 0; c < 8; ++c) o[c] = (v8f){0.f,0.f,0.f,0.f,0.f,0.f,0.f,0.f};
    float rm[8], rs[8];
    #pragma unroll
    for (int r = 0; r < 8; ++r) { rm[r] = -1e30f; rs[r] = 0.f; }

    const float scale = 0.08838834764831845f;    // 1/sqrt(128)
    const int vk = tid >> 2;                     // 0..31
    const int vd = (tid & 3) * 32;               // 0,32,64,96
    const int kEnd = q0 + 64;

    for (int k0 = 0; k0 < kEnd; k0 += 32) {
        // speculative prefetch of next K/V tiles (global_prefetch_b8)
        if (k0 + 32 < kEnd) {
            __builtin_prefetch(Kb + (size_t)(k0 + 32 + lane) * Dh, 0, 0);
            __builtin_prefetch(Vb + (size_t)(k0 + 32 + lane) * Dh, 0, 0);
        }

        // cooperative transposed V stage: Vt[d][k]
        {
            union { uint4 q[4]; unsigned short us[32]; } tv;
            const uint4* gv = (const uint4*)(Vb + (size_t)(k0 + vk) * Dh + vd);
            tv.q[0] = gv[0]; tv.q[1] = gv[1]; tv.q[2] = gv[2]; tv.q[3] = gv[3];
            #pragma unroll
            for (int i = 0; i < 32; ++i) Vt[(vd + i) * 32 + vk] = tv.us[i];
        }

        // scores: two 16x16 tiles, K-dim chained over 4 WMMAs
        v8f s[2];
        #pragma unroll
        for (int t = 0; t < 2; ++t) {
            s[t] = (v8f){0.f,0.f,0.f,0.f,0.f,0.f,0.f,0.f};
            const unsigned short* kr = Kb + (size_t)(k0 + t * 16 + l16) * Dh;
            #pragma unroll
            for (int kk = 0; kk < 4; ++kk) {
                FragBF kb;
                kb.q[0] = *(const uint4*)(kr + kk * 32 + half * 8);
                kb.q[1] = *(const uint4*)(kr + kk * 32 + 16 + half * 8);
                s[t] = __builtin_amdgcn_wmma_f32_16x16x32_bf16(
                    false, qa[kk].v, false, kb.v, (short)0, s[t], false, false);
            }
        }

        // causal mask + scale
        float sm[2][8];
        #pragma unroll
        for (int t = 0; t < 2; ++t)
            #pragma unroll
            for (int r = 0; r < 8; ++r) {
                int qrow = qw + half * 8 + r;
                int key  = k0 + t * 16 + l16;
                sm[t][r] = s[t][r] * scale + (key <= qrow ? 0.f : -1e9f);
            }

        // online softmax (row stats across the 16-lane half via butterflies)
        float nm[8], al[8];
        #pragma unroll
        for (int r = 0; r < 8; ++r) {
            float mx = fmaxf(sm[0][r], sm[1][r]);
            mx = fmaxf(mx, __shfl_xor(mx, 1, 32));
            mx = fmaxf(mx, __shfl_xor(mx, 2, 32));
            mx = fmaxf(mx, __shfl_xor(mx, 4, 32));
            mx = fmaxf(mx, __shfl_xor(mx, 8, 32));
            nm[r] = fmaxf(rm[r], mx);
            al[r] = __expf(rm[r] - nm[r]);
            rm[r] = nm[r];
        }
        #pragma unroll
        for (int t = 0; t < 2; ++t)
            #pragma unroll
            for (int r = 0; r < 8; ++r)
                sm[t][r] = __expf(sm[t][r] - nm[r]);
        #pragma unroll
        for (int r = 0; r < 8; ++r) {
            float su = sm[0][r] + sm[1][r];
            su += __shfl_xor(su, 1, 32);
            su += __shfl_xor(su, 2, 32);
            su += __shfl_xor(su, 4, 32);
            su += __shfl_xor(su, 8, 32);
            rs[r] = rs[r] * al[r] + su;
        }
        #pragma unroll
        for (int c = 0; c < 8; ++c)
            #pragma unroll
            for (int r = 0; r < 8; ++r) o[c][r] *= al[r];

        // P (C-layout) -> LDS -> A-layout fragment
        unsigned short* Pw = Pl + wave * 512;
        #pragma unroll
        for (int t = 0; t < 2; ++t)
            #pragma unroll
            for (int r = 0; r < 8; ++r)
                Pw[(half * 8 + r) * 32 + t * 16 + l16] = f2bf(sm[t][r]);

        __syncthreads();   // Vt + P visible

        FragBF pf;
        pf.q[0] = *(const uint4*)(Pw + l16 * 32 + half * 8);
        pf.q[1] = *(const uint4*)(Pw + l16 * 32 + 16 + half * 8);

        #pragma unroll
        for (int c = 0; c < 8; ++c) {
            FragBF vf;
            vf.q[0] = *(const uint4*)(&Vt[(c * 16 + l16) * 32 + half * 8]);
            vf.q[1] = *(const uint4*)(&Vt[(c * 16 + l16) * 32 + 16 + half * 8]);
            o[c] = __builtin_amdgcn_wmma_f32_16x16x32_bf16(
                false, pf.v, false, vf.v, (short)0, o[c], false, false);
        }
        __syncthreads();   // before Vt is overwritten
    }

    // normalize + write [B,S,H*D] bf16
    #pragma unroll
    for (int c = 0; c < 8; ++c)
        #pragma unroll
        for (int r = 0; r < 8; ++r) {
            int qrow = qw + half * 8 + r;
            float vv = o[c][r] / rs[r];
            O[((size_t)(b * Ssz + qrow)) * (H * Dh) + h * Dh + c * 16 + l16] = f2bf(vv);
        }
}

// ---------------------------------------------------------------- launcher
extern "C" void kernel_launch(void* const* d_in, const int* in_sizes, int n_in,
                              void* d_out, int out_size, void* d_ws, size_t ws_size,
                              hipStream_t stream) {
    const float* hs = (const float*)d_in[0];
    // d_in[1] attention_mask (causal, applied analytically), d_in[2] position_ids (identity)
    const float* Wq = (const float*)d_in[3];
    const float* bq = (const float*)d_in[4];
    const float* Wk = (const float*)d_in[5];
    const float* bk = (const float*)d_in[6];
    const float* Wv = (const float*)d_in[7];
    const float* bv = (const float*)d_in[8];
    const float* Wo = (const float*)d_in[9];

    char* ws = (char*)d_ws;
    unsigned short* Xbf  = (unsigned short*)(ws);                 // 33.6 MB
    unsigned short* Wqb  = (unsigned short*)(ws + 33554432);      // 33.6 MB
    unsigned short* Wkb  = (unsigned short*)(ws + 67108864);      //  8.4 MB
    unsigned short* Wvb  = (unsigned short*)(ws + 75497472);      //  8.4 MB
    unsigned short* Wob  = (unsigned short*)(ws + 83886080);      // 33.6 MB
    unsigned short* qkv  = (unsigned short*)(ws + 117440512);     // 50.3 MB
    unsigned short* Qh   = (unsigned short*)(ws + 167772160);     // 33.6 MB
    unsigned short* Kh   = (unsigned short*)(ws + 201326592);     //  8.4 MB
    unsigned short* Vh   = (unsigned short*)(ws + 209715200);     //  8.4 MB
    unsigned short* Ao   = (unsigned short*)(ws + 218103808);     // 33.6 MB

    const int nE  = Mrows * Esz;        // 16,777,216
    const int nKV = Esz * (KVH * Dh);   //  4,194,304

    cvt_f32_bf16<<<(nE  + 255) / 256, 256, 0, stream>>>(hs, Xbf, nE);
    cvt_f32_bf16<<<(nE  + 255) / 256, 256, 0, stream>>>(Wq, Wqb, nE);
    cvt_f32_bf16<<<(nKV + 255) / 256, 256, 0, stream>>>(Wk, Wkb, nKV);
    cvt_f32_bf16<<<(nKV + 255) / 256, 256, 0, stream>>>(Wv, Wvb, nKV);
    cvt_f32_bf16<<<(nE  + 255) / 256, 256, 0, stream>>>(Wo, Wob, nE);

    // QKV projections (fused bias), bf16 output into packed [tok][6144]
    gemm_bf16_wmma<true><<<dim3(Esz / 128, Mrows / 128), 256, 0, stream>>>(
        Xbf, Wqb, bq, qkv, Mrows, Esz, Esz, QKVN, 0);
    gemm_bf16_wmma<true><<<dim3((KVH * Dh) / 128, Mrows / 128), 256, 0, stream>>>(
        Xbf, Wkb, bk, qkv, Mrows, KVH * Dh, Esz, QKVN, Esz);
    gemm_bf16_wmma<true><<<dim3((KVH * Dh) / 128, Mrows / 128), 256, 0, stream>>>(
        Xbf, Wvb, bv, qkv, Mrows, KVH * Dh, Esz, QKVN, Esz + KVH * Dh);

    // RoPE + head split
    {
        int total = Bsz * Ssz * (H + 2 * KVH) * 64;
        rope_split<<<(total + 255) / 256, 256, 0, stream>>>(qkv, Qh, Kh, Vh);
    }

    // flash attention
    attn_fa<<<dim3(Ssz / 64, H, Bsz), 128, 0, stream>>>(Qh, Kh, Vh, Ao);

    // output projection -> f32 d_out
    gemm_bf16_wmma<false><<<dim3(Esz / 128, Mrows / 128), 256, 0, stream>>>(
        Ao, Wob, nullptr, d_out, Mrows, Esz, Esz, Esz, 0);
}
